// SingleHeadAttention_TA_1786706395420
// MI455X (gfx1250) — matrix operand order repaired and checked by reading, not compiled, not run on hardware
//
#include <hip/hip_runtime.h>
#include <math.h>
#include <stdint.h>

typedef __bf16 bf16;
typedef bf16  v16bf __attribute__((ext_vector_type(16)));
typedef bf16  v8bf  __attribute__((ext_vector_type(8)));
typedef bf16  v4bf  __attribute__((ext_vector_type(4)));
typedef float v8f   __attribute__((ext_vector_type(8)));
typedef int   v4i   __attribute__((ext_vector_type(4)));

#define LDS_STRIDE 40   // 32 bf16 + 8 pad -> 80B rows, conflict-free frag reads

// ---------------- async global->LDS path (CDNA5) ----------------------------
#if __has_builtin(__builtin_amdgcn_global_load_async_to_lds_b128)
#define USE_ASYNC 1
typedef __attribute__((address_space(1))) v4i as1_v4i;
typedef __attribute__((address_space(3))) v4i as3_v4i;
__device__ __forceinline__ void async_b128(const bf16* g, bf16* l) {
  __builtin_amdgcn_global_load_async_to_lds_b128(
      (as1_v4i*)(uintptr_t)g, (as3_v4i*)(uint32_t)(uintptr_t)l, 0, 0);
}
#else
#define USE_ASYNC 0
#endif
__device__ __forceinline__ void wait_async0() {
  asm volatile("s_wait_asynccnt 0x0" ::: "memory");
}
__device__ __forceinline__ void wait_async4() {
  asm volatile("s_wait_asynccnt 0x4" ::: "memory");
}
__device__ __forceinline__ void wait_async8() {
  asm volatile("s_wait_asynccnt 0x8" ::: "memory");
}

// ---------------- fragment loads from LDS (wave32 WMMA bf16 layouts) --------
// A operand 16x32: lane m = lane&15, hf = lane>>4.
// elems 0..7  <- k = hf*8 .. hf*8+7   (VGPR0-3) ; elems 8..15 <- k = 16+hf*8 ..
__device__ __forceinline__ v16bf ld_frag_a(const bf16* sm, int r, int hf) {
  const bf16* p = sm + r * LDS_STRIDE;
  union { v16bf v; v8bf h[2]; } u;
  u.h[0] = *(const v8bf*)(p + hf * 8);
  u.h[1] = *(const v8bf*)(p + 16 + hf * 8);
  return u.v;
}
// B operand 32x16 from LDS storing B^T rows (n-major, k contiguous):
// lane n = lane&15, hf = lane>>4; elems 0..7 <- k = hf*8 .. hf*8+7 ; elems 8..15 <- k = 16+hf*8 ..
__device__ __forceinline__ v16bf ld_frag_b(const bf16* sm, int r, int hf) {
  const bf16* p = sm + r * LDS_STRIDE;
  union { v16bf v; v8bf h[2]; } u;
  u.h[0] = *(const v8bf*)(p + hf * 8);
  u.h[1] = *(const v8bf*)(p + 16 + hf * 8);
  return u.v;
}

__device__ __forceinline__ v8f wmma_bf16(v16bf a, v16bf b, v8f c) {
  return __builtin_amdgcn_wmma_f32_16x16x32_bf16(false, a, false, b, (short)0, c,
                                                 false, false);
}

// 128 rows x 32 cols bf16 tile, 256 threads
__device__ __forceinline__ void load_tile_sync(const bf16* __restrict__ g, int ld,
                                               bf16* sm) {
  int tid = threadIdx.x;
  int quad = tid & 3;
  int row = tid >> 2;
#pragma unroll
  for (int p = 0; p < 2; ++p) {
    int r = row + p * 64;
    *(v4i*)(sm + r * LDS_STRIDE + quad * 8) =
        *(const v4i*)(g + (size_t)r * ld + quad * 8);
  }
}
#if USE_ASYNC
__device__ __forceinline__ void load_tile_async(const bf16* __restrict__ g, int ld,
                                                bf16* sm) {
  int tid = threadIdx.x;
  int quad = tid & 3;
  int row = tid >> 2;
#pragma unroll
  for (int p = 0; p < 2; ++p) {
    int r = row + p * 64;
    async_b128(g + (size_t)r * ld + quad * 8, sm + r * LDS_STRIDE + quad * 8);
  }
}
#endif

// ---------------- elementwise converts --------------------------------------
__global__ __launch_bounds__(256) void k_cvt(const float* __restrict__ in,
                                             bf16* __restrict__ out, long n4) {
  long i = (long)blockIdx.x * blockDim.x + threadIdx.x;
  if (i < n4) {
    float4 x = ((const float4*)in)[i];
    v4bf o;
    o[0] = (bf16)x.x; o[1] = (bf16)x.y; o[2] = (bf16)x.z; o[3] = (bf16)x.w;
    ((v4bf*)out)[i] = o;
  }
}

// W[512,512] f32 -> W^T[512,512] bf16
__global__ void k_wt(const float* __restrict__ W, bf16* __restrict__ Wt) {
  int n = blockIdx.x * 16 + threadIdx.x;
  int kk = blockIdx.y * 16 + threadIdx.y;
  Wt[n * 512 + kk] = (bf16)W[kk * 512 + n];
}

// ---------------- generic bf16 WMMA GEMM ------------------------------------
// C[M,N] = A[M,K] * B  (B given as B^T rows: Bt[n][k], ldb)
// mode 0: bf16 C row-major; mode 1: bf16 C transposed-per-z (C[n*ldc+m], ldc=M);
// mode 2: f32 C row-major + residual add.
__global__ __launch_bounds__(256) void k_gemm(
    const bf16* __restrict__ A, size_t sAz, int lda,
    const bf16* __restrict__ Bt, size_t sBz, int ldb,
    void* __restrict__ Cv, size_t sCz, int ldc,
    const float* __restrict__ R, size_t sRz,
    int K, int mode) {
  __shared__ __align__(16) bf16 smA[2][128 * LDS_STRIDE];
  __shared__ __align__(16) bf16 smB[2][128 * LDS_STRIDE];
  int z = blockIdx.z;
  const bf16* Ab = A + (size_t)z * sAz + (size_t)blockIdx.y * 128 * lda;
  const bf16* Bb = Bt + (size_t)z * sBz + (size_t)blockIdx.x * 128 * ldb;

  int lane = threadIdx.x & 31;
  int wave = threadIdx.x >> 5;
  int wrow = wave >> 1, wcol = wave & 1;   // 4x2 waves -> 128x128 tile
  int nl = lane & 15, hf = lane >> 4;

  v8f acc[2][4] = {};
  int nk = K >> 5;
#if USE_ASYNC
  load_tile_async(Ab, lda, smA[0]);
  load_tile_async(Bb, ldb, smB[0]);
  for (int t = 0; t < nk; ++t) {
    int cur = t & 1;
    if (t + 1 < nk) {
      load_tile_async(Ab + (t + 1) * 32, lda, smA[1 - cur]);
      load_tile_async(Bb + (t + 1) * 32, ldb, smB[1 - cur]);
      wait_async4();            // own 4 loads of tile t complete (in-order)
    } else {
      wait_async0();
    }
    __syncthreads();
    v16bf af[2], bfr[4];
#pragma unroll
    for (int mi = 0; mi < 2; ++mi)
      af[mi] = ld_frag_a(smA[cur], wrow * 32 + mi * 16 + nl, hf);
#pragma unroll
    for (int ni = 0; ni < 4; ++ni)
      bfr[ni] = ld_frag_b(smB[cur], wcol * 64 + ni * 16 + nl, hf);
#pragma unroll
    for (int mi = 0; mi < 2; ++mi)
#pragma unroll
      for (int ni = 0; ni < 4; ++ni)
        acc[mi][ni] = wmma_bf16(af[mi], bfr[ni], acc[mi][ni]);
    __syncthreads();
  }
#else
  for (int t = 0; t < nk; ++t) {
    load_tile_sync(Ab + t * 32, lda, smA[0]);
    load_tile_sync(Bb + t * 32, ldb, smB[0]);
    __syncthreads();
    v16bf af[2], bfr[4];
#pragma unroll
    for (int mi = 0; mi < 2; ++mi)
      af[mi] = ld_frag_a(smA[0], wrow * 32 + mi * 16 + nl, hf);
#pragma unroll
    for (int ni = 0; ni < 4; ++ni)
      bfr[ni] = ld_frag_b(smB[0], wcol * 64 + ni * 16 + nl, hf);
#pragma unroll
    for (int mi = 0; mi < 2; ++mi)
#pragma unroll
      for (int ni = 0; ni < 4; ++ni)
        acc[mi][ni] = wmma_bf16(af[mi], bfr[ni], acc[mi][ni]);
    __syncthreads();
  }
#endif

  // 32-bit epilogue addressing: fold z-stride into base, step with small ints
  int m0 = blockIdx.y * 128 + wrow * 32 + hf * 8;
  int n0 = blockIdx.x * 128 + wcol * 64 + nl;
  if (mode == 0) {
    bf16* C = (bf16*)Cv + (size_t)z * sCz + (size_t)m0 * ldc + n0;
#pragma unroll
    for (int mi = 0; mi < 2; ++mi) {
      bf16* Cm = C + mi * 16 * ldc;
#pragma unroll
      for (int vv = 0; vv < 8; ++vv) {
        bf16* Cr = Cm + vv * ldc;
#pragma unroll
        for (int ni = 0; ni < 4; ++ni) Cr[ni * 16] = (bf16)acc[mi][ni][vv];
      }
    }
  } else if (mode == 1) {
    bf16* C = (bf16*)Cv + (size_t)z * sCz + (size_t)n0 * ldc + m0;
#pragma unroll
    for (int ni = 0; ni < 4; ++ni) {
      bf16* Cn = C + ni * 16 * ldc;
#pragma unroll
      for (int mi = 0; mi < 2; ++mi)
#pragma unroll
        for (int vv = 0; vv < 8; ++vv)
          Cn[mi * 16 + vv] = (bf16)acc[mi][ni][vv];
    }
  } else {
    float* C = (float*)Cv + (size_t)z * sCz + (size_t)m0 * ldc + n0;
    const float* Rb = R + (size_t)z * sRz + (size_t)m0 * ldc + n0;
#pragma unroll
    for (int mi = 0; mi < 2; ++mi) {
      float* Cm = C + mi * 16 * ldc;
      const float* Rm = Rb + mi * 16 * ldc;
#pragma unroll
      for (int vv = 0; vv < 8; ++vv) {
        float* Cr = Cm + vv * ldc;
        const float* Rr = Rm + vv * ldc;
#pragma unroll
        for (int ni = 0; ni < 4; ++ni)
          Cr[ni * 16] = acc[mi][ni][vv] + Rr[ni * 16];
      }
    }
  }
}

// ---------------- fused attention-score kernel ------------------------------
// S = (qp.kp^T + q2.k2^T) * scale ; S *= mask ; S = tanh(S) ; S *= mask
__global__ __launch_bounds__(256) void k_attn(
    const bf16* __restrict__ qp, const bf16* __restrict__ kp,
    const bf16* __restrict__ q2, const bf16* __restrict__ k2,
    const int* __restrict__ lens, float* __restrict__ attnf,
    bf16* __restrict__ attnb, float scale) {
  const int L = 2048, D = 512;
  __shared__ __align__(16) bf16 smA1[2][128 * LDS_STRIDE];
  __shared__ __align__(16) bf16 smB1[2][128 * LDS_STRIDE];
  __shared__ __align__(16) bf16 smA2[2][128 * LDS_STRIDE];
  __shared__ __align__(16) bf16 smB2[2][128 * LDS_STRIDE];
  size_t zo = (size_t)blockIdx.z * L * D;
  const bf16* Aq = qp + zo + (size_t)blockIdx.y * 128 * D;
  const bf16* Bk = kp + zo + (size_t)blockIdx.x * 128 * D;  // B^T rows = key rows
  const bf16* A2 = q2 + zo + (size_t)blockIdx.y * 128 * D;
  const bf16* B2 = k2 + zo + (size_t)blockIdx.x * 128 * D;

  int lane = threadIdx.x & 31;
  int wave = threadIdx.x >> 5;
  int wrow = wave >> 1, wcol = wave & 1;
  int nl = lane & 15, hf = lane >> 4;

  v8f acc[2][4] = {};
  const int nk = D >> 5;  // 16
#if USE_ASYNC
  load_tile_async(Aq, D, smA1[0]);
  load_tile_async(Bk, D, smB1[0]);
  load_tile_async(A2, D, smA2[0]);
  load_tile_async(B2, D, smB2[0]);
#endif
  for (int t = 0; t < nk; ++t) {
    int cur = t & 1;
#if USE_ASYNC
    if (t + 1 < nk) {
      int k1 = (t + 1) * 32;
      load_tile_async(Aq + k1, D, smA1[1 - cur]);
      load_tile_async(Bk + k1, D, smB1[1 - cur]);
      load_tile_async(A2 + k1, D, smA2[1 - cur]);
      load_tile_async(B2 + k1, D, smB2[1 - cur]);
      wait_async8();
    } else {
      wait_async0();
    }
#else
    cur = 0;
    load_tile_sync(Aq + t * 32, D, smA1[0]);
    load_tile_sync(Bk + t * 32, D, smB1[0]);
    load_tile_sync(A2 + t * 32, D, smA2[0]);
    load_tile_sync(B2 + t * 32, D, smB2[0]);
#endif
    __syncthreads();
    {  // product 1
      v16bf af[2], bfr[4];
#pragma unroll
      for (int mi = 0; mi < 2; ++mi)
        af[mi] = ld_frag_a(smA1[cur], wrow * 32 + mi * 16 + nl, hf);
#pragma unroll
      for (int ni = 0; ni < 4; ++ni)
        bfr[ni] = ld_frag_b(smB1[cur], wcol * 64 + ni * 16 + nl, hf);
#pragma unroll
      for (int mi = 0; mi < 2; ++mi)
#pragma unroll
        for (int ni = 0; ni < 4; ++ni)
          acc[mi][ni] = wmma_bf16(af[mi], bfr[ni], acc[mi][ni]);
    }
    {  // product 2 accumulates into the same C
      v16bf af[2], bfr[4];
#pragma unroll
      for (int mi = 0; mi < 2; ++mi)
        af[mi] = ld_frag_a(smA2[cur], wrow * 32 + mi * 16 + nl, hf);
#pragma unroll
      for (int ni = 0; ni < 4; ++ni)
        bfr[ni] = ld_frag_b(smB2[cur], wcol * 64 + ni * 16 + nl, hf);
#pragma unroll
      for (int mi = 0; mi < 2; ++mi)
#pragma unroll
        for (int ni = 0; ni < 4; ++ni)
          acc[mi][ni] = wmma_bf16(af[mi], bfr[ni], acc[mi][ni]);
    }
    __syncthreads();
  }

  int len = lens[blockIdx.z];
  int m0 = blockIdx.y * 128 + wrow * 32 + hf * 8;
  int n0 = blockIdx.x * 128 + wcol * 64 + nl;
  size_t base = (size_t)blockIdx.z * L * L + (size_t)m0 * L + n0;
  float* fp = attnf + base;
  bf16*  bp = attnb + base;
#pragma unroll
  for (int ni = 0; ni < 4; ++ni) {
    int col = n0 + ni * 16;
    float mv = (col < len) ? 1.0f : 0.0f;
#pragma unroll
    for (int mi = 0; mi < 2; ++mi)
#pragma unroll
      for (int vv = 0; vv < 8; ++vv) {
        int off = (mi * 16 + vv) * L + ni * 16;  // compile-time L -> immediates
        float s = acc[mi][ni][vv] * scale * mv;
        float t = tanhf(s) * mv;
        fp[off] = t;
        bp[off] = (bf16)t;
      }
  }
}

// ---------------- layernorm over D=512 --------------------------------------
__global__ __launch_bounds__(128) void k_ln(const float* __restrict__ x,
                                            const float* __restrict__ g,
                                            const float* __restrict__ b,
                                            float* __restrict__ out) {
  int row = blockIdx.x;
  int t = threadIdx.x;
  const float* xr = x + (size_t)row * 512;
  float4 v = ((const float4*)xr)[t];
  float s = v.x + v.y + v.z + v.w;
  float s2 = v.x * v.x + v.y * v.y + v.z * v.z + v.w * v.w;
  __shared__ float sm1[128];
  __shared__ float sm2[128];
  sm1[t] = s; sm2[t] = s2;
  __syncthreads();
  for (int off = 64; off > 0; off >>= 1) {
    if (t < off) { sm1[t] += sm1[t + off]; sm2[t] += sm2[t + off]; }
    __syncthreads();
  }
  float mu = sm1[0] * (1.0f / 512.0f);
  float var = sm2[0] * (1.0f / 512.0f) - mu * mu;
  float inv = rsqrtf(var + 1e-6f);
  float4 gg = ((const float4*)g)[t];
  float4 bb = ((const float4*)b)[t];
  float* orow = out + (size_t)row * 512;
  float4 o;
  o.x = (v.x - mu) * inv * gg.x + bb.x;
  o.y = (v.y - mu) * inv * gg.y + bb.y;
  o.z = (v.z - mu) * inv * gg.z + bb.z;
  o.w = (v.w - mu) * inv * gg.w + bb.w;
  ((float4*)orow)[t] = o;
}

// ---------------- host-side orchestration -----------------------------------
extern "C" void kernel_launch(void* const* d_in, const int* in_sizes, int n_in,
                              void* d_out, int out_size, void* d_ws, size_t ws_size,
                              hipStream_t stream) {
  (void)in_sizes; (void)n_in; (void)out_size; (void)ws_size;
  const float* qin  = (const float*)d_in[0];
  const float* kin  = (const float*)d_in[1];
  const float* vin  = (const float*)d_in[2];
  const int*   lens = (const int*)d_in[3];
  const float* dist = (const float*)d_in[4];
  const float* Wq   = (const float*)d_in[5];
  const float* Wk   = (const float*)d_in[6];
  const float* Wv   = (const float*)d_in[7];
  const float* Wq2  = (const float*)d_in[8];
  const float* Wk2  = (const float*)d_in[9];
  const float* Wfc  = (const float*)d_in[10];
  const float* lng  = (const float*)d_in[11];
  const float* lnb  = (const float*)d_in[12];

  const size_t Bn = 8, L = 2048, D = 512;
  const size_t SP = Bn * L * D;   // 8,388,608
  const size_t SD = Bn * L * L;   // 33,554,432
  const size_t WSZ = D * D;

  bf16* qbf    = (bf16*)d_ws;
  bf16* kbf    = qbf + SP;
  bf16* vbf    = kbf + SP;
  bf16* distbf = vbf + SP;
  bf16* wt0    = distbf + SD;       // Wq^T
  bf16* wt1    = wt0 + WSZ;         // Wk^T
  bf16* wt2    = wt1 + WSZ;         // Wv^T
  bf16* wt3    = wt2 + WSZ;         // Wq2^T
  bf16* wt4    = wt3 + WSZ;         // Wk2^T
  bf16* wt5    = wt4 + WSZ;         // Wfc^T
  bf16* qpb    = wt5 + WSZ;         // qp  [b][l][d]
  bf16* kpb    = qpb + SP;          // kp  [b][l][d]
  bf16* vpt    = kpb + SP;          // vp^T [b][d][l]
  bf16* qct    = vpt + SP;          // (q@Wq2)^T [b][d][l]
  bf16* kct    = qct + SP;          // (k@Wk2)^T [b][d][l]
  bf16* q2b    = kct + SP;          // q2 [b][l][d]
  bf16* k2b    = q2b + SP;          // k2 [b][l][d]
  // reuse regions once their originals are dead:
  bf16*  attnbf = distbf;           // bf16 copy of attn (dist dead after q2/k2)
  bf16*  outpre = vbf;              // attn @ vp         (vbf dead after proj)
  float* outlin = (float*)d_ws;     // fc output+residual (qbf/kbf dead)

  float* out_main = (float*)d_out;        // [B,L,D]
  float* out_attn = out_main + SP;        // [B,L,L]

  dim3 blk(256);

  // 1) converts
  k_cvt<<<(unsigned)((SP / 4 + 255) / 256), 256, 0, stream>>>(qin, qbf, (long)(SP / 4));
  k_cvt<<<(unsigned)((SP / 4 + 255) / 256), 256, 0, stream>>>(kin, kbf, (long)(SP / 4));
  k_cvt<<<(unsigned)((SP / 4 + 255) / 256), 256, 0, stream>>>(vin, vbf, (long)(SP / 4));
  k_cvt<<<(unsigned)((SD / 4 + 255) / 256), 256, 0, stream>>>(dist, distbf, (long)(SD / 4));

  // 2) weight transposes f32 -> bf16
  dim3 wtg(32, 32), wtb(16, 16);
  k_wt<<<wtg, wtb, 0, stream>>>(Wq,  wt0);
  k_wt<<<wtg, wtb, 0, stream>>>(Wk,  wt1);
  k_wt<<<wtg, wtb, 0, stream>>>(Wv,  wt2);
  k_wt<<<wtg, wtb, 0, stream>>>(Wq2, wt3);
  k_wt<<<wtg, wtb, 0, stream>>>(Wk2, wt4);
  k_wt<<<wtg, wtb, 0, stream>>>(Wfc, wt5);

  // 3) projections: [2048x512] x [512x512] per batch (z=8)
  dim3 pg(4, 16, 8);
  k_gemm<<<pg, blk, 0, stream>>>(qbf, L * D, (int)D, wt0, 0, (int)D,
                                 qpb, L * D, (int)D, nullptr, 0, (int)D, 0);
  k_gemm<<<pg, blk, 0, stream>>>(kbf, L * D, (int)D, wt1, 0, (int)D,
                                 kpb, L * D, (int)D, nullptr, 0, (int)D, 0);
  k_gemm<<<pg, blk, 0, stream>>>(vbf, L * D, (int)D, wt2, 0, (int)D,
                                 vpt, D * L, (int)L, nullptr, 0, (int)D, 1);
  k_gemm<<<pg, blk, 0, stream>>>(qbf, L * D, (int)D, wt3, 0, (int)D,
                                 qct, D * L, (int)L, nullptr, 0, (int)D, 1);
  k_gemm<<<pg, blk, 0, stream>>>(kbf, L * D, (int)D, wt4, 0, (int)D,
                                 kct, D * L, (int)L, nullptr, 0, (int)D, 1);

  // 4) time-aware reweighting: q2 = dist @ (q@Wq2), k2 = dist @ (k@Wk2)
  k_gemm<<<pg, blk, 0, stream>>>(distbf, L * L, (int)L, qct, D * L, (int)L,
                                 q2b, L * D, (int)D, nullptr, 0, (int)L, 0);
  k_gemm<<<pg, blk, 0, stream>>>(distbf, L * L, (int)L, kct, D * L, (int)L,
                                 k2b, L * D, (int)D, nullptr, 0, (int)L, 0);

  // 5) fused scores: (qp.kp^T + q2.k2^T)/sqrt(D), mask, tanh, mask
  dim3 ag(16, 16, 8);
  k_attn<<<ag, blk, 0, stream>>>(qpb, kpb, q2b, k2b, lens, out_attn, attnbf,
                                 0.04419417382415922f);  // 1/sqrt(512)

  // 6) out_pre = attn @ vp
  k_gemm<<<pg, blk, 0, stream>>>(attnbf, L * L, (int)L, vpt, D * L, (int)L,
                                 outpre, L * D, (int)D, nullptr, 0, (int)L, 0);

  // 7) out_lin = out_pre @ Wfc + residual(q)
  dim3 fg(4, 128, 1);
  k_gemm<<<fg, blk, 0, stream>>>(outpre, 0, (int)D, wt5, 0, (int)D,
                                 outlin, 0, (int)D, qin, 0, (int)D, 2);

  // 8) layernorm -> d_out
  k_ln<<<(unsigned)(Bn * L), 128, 0, stream>>>(outlin, lng, lnb, out_main);
}
